// ViT_UNet_25340307047021
// MI455X (gfx1250) — matrix operand order repaired and checked by reading, not compiled, not run on hardware
//
#include <hip/hip_runtime.h>
#include <math.h>

// ---------------- dims ----------------
#define B_   4
#define N_   1024
#define D_   768
#define H_   8
#define HD_  96
#define HID_ 3072

typedef unsigned short u16;
typedef unsigned int   u32;
typedef __attribute__((ext_vector_type(16))) __bf16 v16bf;
typedef __attribute__((ext_vector_type(8)))  float  v8f;
typedef __attribute__((ext_vector_type(4)))  unsigned int u32x4;

// ---------------- bf16 helpers ----------------
__device__ __forceinline__ u16 f2bf(float f) {
  u32 u = __float_as_uint(f);
  u32 r = u + 0x7FFFu + ((u >> 16) & 1u);   // round-to-nearest-even
  return (u16)(r >> 16);
}
__device__ __forceinline__ float bf2f(u16 h) {
  return __uint_as_float(((u32)h) << 16);
}

// ---------------- WMMA fragment loaders (CDNA5 16-bit layouts) ----------------
// A 16x32: lane row = lane&15; halves 0..7 -> K = k0 + (lane>>4)*8 + (0..7)
//                    halves 8..15 -> K = k0 + 16 + (lane>>4)*8 + (0..7)
__device__ __forceinline__ v16bf load_a(const u16* base, int ld, int lane, int k0) {
  const u16* p = base + (size_t)(lane & 15) * ld + k0 + ((lane >> 4) << 3);
  union { u32x4 u[2]; v16bf v; } un;
  un.u[0] = *(const u32x4*)(p);
  un.u[1] = *(const u32x4*)(p + 16);
  return un.v;
}
// B 32x16: lane col = lane&15; halves 0..7 -> K = k0 + (lane>>4)*8 + (0..7)
//                    halves 8..15 -> K = k0 + 16 + (lane>>4)*8 + (0..7)
__device__ __forceinline__ v16bf load_b(const u16* base, int ld, int lane, int k0) {
  const u16* p = base + (size_t)(lane & 15) * ld + k0 + ((lane >> 4) << 3);
  union { u32x4 u[2]; v16bf v; } un;
  un.u[0] = *(const u32x4*)(p);
  un.u[1] = *(const u32x4*)(p + 16);
  return un.v;
}
__device__ __forceinline__ v8f wmma_bf16(v16bf a, v16bf b, v8f c) {
  return __builtin_amdgcn_wmma_f32_16x16x32_bf16(false, a, false, b, (short)0, c,
                                                 false, false);
}

// ---------------- stats zero ----------------
__global__ void k_zero_stats(double* stats) {
  int i = threadIdx.x;
  if (i < 64) stats[i] = 0.0;
}

// ---------------- fp32 -> bf16 weight convert ----------------
__global__ void k_cvt(const float* __restrict__ src, u16* __restrict__ dst, int n) {
  int i = blockIdx.x * 256 + threadIdx.x;
  if (i < n) dst[i] = f2bf(src[i]);
}

// ---------------- per-token 3x3 conv QKV ----------------
__global__ void k_qkv_conv(const float* __restrict__ x,
                           const float* __restrict__ qw,
                           const float* __restrict__ kw,
                           const float* __restrict__ vw,
                           u16* __restrict__ qb, u16* __restrict__ kb,
                           u16* __restrict__ vtb) {
  int token = blockIdx.x;               // b*N + n
  int b = token >> 10, n = token & 1023;
  __shared__ float xt[768];
  __shared__ float cw[3][81];
  int t = threadIdx.x;
  for (int i = t; i < 768; i += 256) xt[i] = x[(size_t)token * 768 + i];
  if (t < 243) {
    int which = t / 81, idx = t - which * 81;
    const float* src = (which == 0) ? qw : ((which == 1) ? kw : vw);
    cw[which][idx] = src[idx];
  }
  __syncthreads();
  for (int e = t; e < 768; e += 256) {
    int co = e >> 8, ij = e & 255, i = ij >> 4, j = ij & 15;
    float a0 = 0.f, a1 = 0.f, a2 = 0.f;
    for (int ci = 0; ci < 3; ci++) {
      for (int ki = 0; ki < 3; ki++) {
        int ii = i + ki - 1;
        if (ii < 0 || ii > 15) continue;
        for (int kj = 0; kj < 3; kj++) {
          int jj = j + kj - 1;
          if (jj < 0 || jj > 15) continue;
          float xv = xt[ci * 256 + ii * 16 + jj];
          int widx = co * 27 + ci * 9 + ki * 3 + kj;
          a0 += xv * cw[0][widx];
          a1 += xv * cw[1][widx];
          a2 += xv * cw[2][widx];
        }
      }
    }
    int h = e / 96, d = e - h * 96;
    size_t bh = (size_t)b * H_ + h;
    qb[(bh * N_ + n) * HD_ + d] = f2bf(a0);
    kb[(bh * N_ + n) * HD_ + d] = f2bf(a1);
    vtb[(bh * HD_ + d) * N_ + n] = f2bf(a2);
  }
}

// ---------------- colsum over v (for BN fold) ----------------
__global__ void k_colsum(const u16* __restrict__ vtb, float* __restrict__ colsum) {
  int row = (blockIdx.x * blockDim.x + threadIdx.x) >> 5;  // (b*8+g)*96+d
  int lane = threadIdx.x & 31;
  if (row >= B_ * H_ * HD_) return;
  const u16* p = vtb + (size_t)row * N_;
  float s = 0.f;
  for (int i = lane; i < N_; i += 32) s += bf2f(p[i]);
  for (int m = 1; m < 32; m <<= 1) s += __shfl_xor(s, m, 32);
  if (lane == 0) colsum[row] = s;
}

// ---------------- fused QK^T + softmax -> bf16 attn ----------------
__global__ void k_qk_softmax(const u16* __restrict__ qb, const u16* __restrict__ kb,
                             u16* __restrict__ attn) {
  int blk = blockIdx.x;
  int nt = blk & 63;          // 16-row tile in N
  int bh = blk >> 6;          // b*8+h
  int wave = threadIdx.x >> 5, lane = threadIdx.x & 31;
  const u16* qbase = qb + ((size_t)bh * N_ + nt * 16) * HD_;
  const u16* kbase = kb + (size_t)bh * N_ * HD_;
  v16bf a0 = load_a(qbase, HD_, lane, 0);
  v16bf a1 = load_a(qbase, HD_, lane, 32);
  v16bf a2 = load_a(qbase, HD_, lane, 64);
  v8f acc[8];
  for (int ct = 0; ct < 8; ct++) {
    const u16* bb = kbase + (size_t)(wave * 128 + ct * 16) * HD_;
    v8f c = {};
    c = wmma_bf16(a0, load_b(bb, HD_, lane, 0), c);
    c = wmma_bf16(a1, load_b(bb, HD_, lane, 32), c);
    c = wmma_bf16(a2, load_b(bb, HD_, lane, 64), c);
    acc[ct] = c;
  }
  const float scale = 0.1020620726159658f;  // 1/sqrt(96)
  for (int ct = 0; ct < 8; ct++)
    for (int r = 0; r < 8; r++) acc[ct][r] *= scale;

  __shared__ float redA[8][16];
  __shared__ float redB[8][16];
  int rowoff = (lane >= 16) ? 8 : 0;
  // row max (this wave's 128 cols)
  float rmax[8];
  for (int r = 0; r < 8; r++) {
    float m = acc[0][r];
    for (int ct = 1; ct < 8; ct++) m = fmaxf(m, acc[ct][r]);
    for (int d = 1; d < 16; d <<= 1) m = fmaxf(m, __shfl_xor(m, d, 32));
    rmax[r] = m;
  }
  if ((lane & 15) == 0)
    for (int r = 0; r < 8; r++) redA[wave][rowoff + r] = rmax[r];
  __syncthreads();
  float gmax[8];
  for (int r = 0; r < 8; r++) {
    float m = redA[0][rowoff + r];
    for (int w = 1; w < 8; w++) m = fmaxf(m, redA[w][rowoff + r]);
    gmax[r] = m;
  }
  // exp + row sum
  float rsum[8];
  for (int r = 0; r < 8; r++) {
    float s = 0.f;
    for (int ct = 0; ct < 8; ct++) {
      float e = __expf(acc[ct][r] - gmax[r]);
      acc[ct][r] = e;
      s += e;
    }
    for (int d = 1; d < 16; d <<= 1) s += __shfl_xor(s, d, 32);
    rsum[r] = s;
  }
  if ((lane & 15) == 0)
    for (int r = 0; r < 8; r++) redB[wave][rowoff + r] = rsum[r];
  __syncthreads();
  for (int r = 0; r < 8; r++) {
    float s = 0.f;
    for (int w = 0; w < 8; w++) s += redB[w][rowoff + r];
    float inv = 1.0f / s;
    int n = nt * 16 + rowoff + r;
    size_t rowbase = ((size_t)bh * N_ + n) * N_;
    for (int ct = 0; ct < 8; ct++) {
      int m = wave * 128 + ct * 16 + (lane & 15);
      attn[rowbase + m] = f2bf(acc[ct][r] * inv);
    }
  }
}

// ---------------- in-place 8x8 head mix + BN batch stats (fp64 atomics) ----------
__global__ void k_mix_stats(u16* __restrict__ attn, const float* __restrict__ rw,
                            const float* __restrict__ rb, double* __restrict__ stats) {
  __shared__ float srw[64];
  __shared__ float srb[8];
  int t = threadIdx.x;
  if (t < 64) srw[t] = rw[t];
  if (t < 8) srb[t] = rb[t];
  __syncthreads();
  size_t p = (size_t)blockIdx.x * 256 + t;    // over B*N*N
  int b = (int)(p >> 20);
  size_t nm = p & 1048575;
  u16* base = attn + (size_t)b * 8 * 1048576 + nm;
  float av[8], ov[8];
  for (int h = 0; h < 8; h++) av[h] = bf2f(base[(size_t)h * 1048576]);
  for (int g = 0; g < 8; g++) {
    float o = srb[g];
    for (int h = 0; h < 8; h++) o += srw[g * 8 + h] * av[h];
    ov[g] = o;
  }
  for (int g = 0; g < 8; g++) base[(size_t)g * 1048576] = f2bf(ov[g]);
  int lane = t & 31;
  for (int g = 0; g < 8; g++) {
    float s = ov[g], ss = ov[g] * ov[g];
    for (int d = 1; d < 32; d <<= 1) {
      s += __shfl_xor(s, d, 32);
      ss += __shfl_xor(ss, d, 32);
    }
    if (lane == 0) {
      atomicAdd(&stats[g], (double)s);
      atomicAdd(&stats[8 + g], (double)ss);
    }
  }
}

__global__ void k_bn_finalize(const double* __restrict__ stats,
                              const float* __restrict__ bn_g,
                              const float* __restrict__ bn_b,
                              float* __restrict__ coef) {
  int g = threadIdx.x;
  if (g >= 8) return;
  double cnt = 4194304.0;  // B*N*N
  double mu = stats[g] / cnt;
  double var = stats[8 + g] / cnt - mu * mu;
  float rs = rsqrtf((float)var + 1e-5f);
  float s = bn_g[g] * rs;
  float tt = bn_b[g] - (float)mu * s;
  coef[g] = s;
  coef[8 + g] = tt;
}

// ---------------- attn @ v with folded-BN epilogue -> bf16 ctx [B,N,D] ----------
__global__ void k_attnv(const u16* __restrict__ attn, const u16* __restrict__ vtb,
                        const float* __restrict__ coef, const float* __restrict__ colsum,
                        u16* __restrict__ ctx) {
  int blk = blockIdx.x;
  int rt = blk & 7;           // 128-row group
  int bh = blk >> 3;          // b*8+g
  int b = bh >> 3, g = bh & 7;
  int wave = threadIdx.x >> 5, lane = threadIdx.x & 31;
  int row0 = rt * 128 + wave * 16;
  const u16* abase = attn + ((size_t)bh * N_ + row0) * N_;
  const u16* vbase = vtb + (size_t)bh * HD_ * N_;
  v8f acc[6] = {{}, {}, {}, {}, {}, {}};
  for (int k0 = 0; k0 < N_; k0 += 32) {
    v16bf a = load_a(abase, N_, lane, k0);
    for (int ct = 0; ct < 6; ct++) {
      v16bf bf = load_b(vbase + (size_t)(ct * 16) * N_, N_, lane, k0);
      acc[ct] = wmma_bf16(a, bf, acc[ct]);
    }
  }
  float s = coef[g], tt = coef[8 + g];
  int nc = lane & 15, ro = (lane >= 16) ? 8 : 0;
  for (int ct = 0; ct < 6; ct++) {
    int d = ct * 16 + nc;
    float cs = colsum[bh * HD_ + d];
    for (int r = 0; r < 8; r++) {
      int n = row0 + ro + r;
      float o = s * acc[ct][r] + tt * cs;
      ctx[((size_t)b * N_ + n) * D_ + g * HD_ + d] = f2bf(o);
    }
  }
}

// ---------------- generic NT WMMA mainloop: wave -> 16x32 tile -----------------
__device__ __forceinline__ void gemm_nt(const u16* __restrict__ A,
                                        const u16* __restrict__ Bm, int K, int lane,
                                        v8f& c0, v8f& c1) {
  const u16* B1 = Bm + (size_t)16 * K;
  for (int k0 = 0; k0 < K; k0 += 32) {
    v16bf a = load_a(A, K, lane, k0);
    v16bf b0 = load_b(Bm, K, lane, k0);
    v16bf b1 = load_b(B1, K, lane, k0);
    c0 = wmma_bf16(a, b0, c0);
    c1 = wmma_bf16(a, b1, c1);
  }
}

// ---------------- proj: ctx @ proj_w^T + b + x -> r1, LN1 stats ----------------
__global__ void k_proj(const u16* __restrict__ ctx, const u16* __restrict__ pw,
                       const float* __restrict__ pb, const float* __restrict__ x,
                       float* __restrict__ r1, double* __restrict__ stats) {
  int wave = threadIdx.x >> 5, lane = threadIdx.x & 31;
  int row0 = blockIdx.x * 64 + (wave >> 1) * 16;
  int col0 = blockIdx.y * 64 + (wave & 1) * 32;
  v8f c0 = {}, c1 = {};
  gemm_nt(ctx + (size_t)row0 * D_, pw + (size_t)col0 * D_, D_, lane, c0, c1);
  int nc = lane & 15, ro = (lane >= 16) ? 8 : 0;
  float s = 0.f, ss = 0.f;
  for (int half = 0; half < 2; half++) {
    int col = col0 + half * 16 + nc;
    float bias = pb[col];
    for (int r = 0; r < 8; r++) {
      int row = row0 + ro + r;
      float v = (half ? c1[r] : c0[r]) + bias + x[(size_t)row * D_ + col];
      r1[(size_t)row * D_ + col] = v;
      s += v;
      ss += v * v;
    }
  }
  for (int d = 1; d < 32; d <<= 1) {
    s += __shfl_xor(s, d, 32);
    ss += __shfl_xor(ss, d, 32);
  }
  if (lane == 0) {
    int b = row0 >> 10;
    atomicAdd(&stats[16 + b], (double)s);
    atomicAdd(&stats[20 + b], (double)ss);
  }
}

__global__ void k_ln_finalize(const double* __restrict__ sum,
                              const double* __restrict__ sumsq,
                              float* __restrict__ coef) {
  int b = threadIdx.x;
  if (b >= 4) return;
  double cnt = 786432.0;  // N*D
  double mu = sum[b] / cnt;
  double var = sumsq[b] / cnt - mu * mu;
  coef[2 * b] = (float)mu;
  coef[2 * b + 1] = rsqrtf((float)var + 1e-5f);
}

__global__ void k_ln1_apply(const float* __restrict__ r1, const float* __restrict__ coef,
                            const float* __restrict__ g, const float* __restrict__ bb,
                            float* __restrict__ x1f, u16* __restrict__ x1b) {
  size_t i = (size_t)blockIdx.x * 256 + threadIdx.x;
  int b = (int)(i / 786432);
  size_t nd = i - (size_t)b * 786432;
  float v = (r1[i] - coef[2 * b]) * coef[2 * b + 1] * g[nd] + bb[nd];
  x1f[i] = v;
  x1b[i] = f2bf(v);
}

// ---------------- ff1: x1 @ w1^T + b1 -> exact GELU -> bf16 h ------------------
__global__ void k_ff1(const u16* __restrict__ x1b, const u16* __restrict__ w1,
                      const float* __restrict__ b1, u16* __restrict__ hb) {
  int wave = threadIdx.x >> 5, lane = threadIdx.x & 31;
  int row0 = blockIdx.x * 64 + (wave >> 1) * 16;
  int col0 = blockIdx.y * 64 + (wave & 1) * 32;
  v8f c0 = {}, c1 = {};
  gemm_nt(x1b + (size_t)row0 * D_, w1 + (size_t)col0 * D_, D_, lane, c0, c1);
  int nc = lane & 15, ro = (lane >= 16) ? 8 : 0;
  for (int half = 0; half < 2; half++) {
    int col = col0 + half * 16 + nc;
    float bias = b1[col];
    for (int r = 0; r < 8; r++) {
      int row = row0 + ro + r;
      float v = (half ? c1[r] : c0[r]) + bias;
      float ge = 0.5f * v * (1.0f + erff(v * 0.70710678118654752f));
      hb[(size_t)row * HID_ + col] = f2bf(ge);
    }
  }
}

// ---------------- ff2: h @ w2^T + b2 + x1 -> r2, LN2 stats ---------------------
__global__ void k_ff2(const u16* __restrict__ hb, const u16* __restrict__ w2,
                      const float* __restrict__ b2, const float* __restrict__ x1f,
                      float* __restrict__ r2, double* __restrict__ stats) {
  int wave = threadIdx.x >> 5, lane = threadIdx.x & 31;
  int row0 = blockIdx.x * 64 + (wave >> 1) * 16;
  int col0 = blockIdx.y * 64 + (wave & 1) * 32;
  v8f c0 = {}, c1 = {};
  gemm_nt(hb + (size_t)row0 * HID_, w2 + (size_t)col0 * HID_, HID_, lane, c0, c1);
  int nc = lane & 15, ro = (lane >= 16) ? 8 : 0;
  float s = 0.f, ss = 0.f;
  for (int half = 0; half < 2; half++) {
    int col = col0 + half * 16 + nc;
    float bias = b2[col];
    for (int r = 0; r < 8; r++) {
      int row = row0 + ro + r;
      float v = (half ? c1[r] : c0[r]) + bias + x1f[(size_t)row * D_ + col];
      r2[(size_t)row * D_ + col] = v;
      s += v;
      ss += v * v;
    }
  }
  for (int d = 1; d < 32; d <<= 1) {
    s += __shfl_xor(s, d, 32);
    ss += __shfl_xor(ss, d, 32);
  }
  if (lane == 0) {
    int b = row0 >> 10;
    atomicAdd(&stats[24 + b], (double)s);
    atomicAdd(&stats[28 + b], (double)ss);
  }
}

__global__ void k_ln2_apply(const float* __restrict__ r2, const float* __restrict__ coef,
                            const float* __restrict__ g, const float* __restrict__ bb,
                            float* __restrict__ out) {
  size_t i = (size_t)blockIdx.x * 256 + threadIdx.x;
  int b = (int)(i / 786432);
  size_t nd = i - (size_t)b * 786432;
  out[i] = (r2[i] - coef[2 * b]) * coef[2 * b + 1] * g[nd] + bb[nd];
}

// =================================================================================
extern "C" void kernel_launch(void* const* d_in, const int* in_sizes, int n_in,
                              void* d_out, int out_size, void* d_ws, size_t ws_size,
                              hipStream_t stream) {
  (void)in_sizes; (void)n_in; (void)out_size; (void)ws_size;
  const float* x   = (const float*)d_in[0];
  const float* qw  = (const float*)d_in[1];
  const float* kw  = (const float*)d_in[2];
  const float* vw  = (const float*)d_in[3];
  const float* rw  = (const float*)d_in[4];
  const float* rb  = (const float*)d_in[5];
  const float* bng = (const float*)d_in[6];
  const float* bnb = (const float*)d_in[7];
  const float* pw  = (const float*)d_in[8];
  const float* pb  = (const float*)d_in[9];
  const float* lng = (const float*)d_in[10];
  const float* lnb = (const float*)d_in[11];
  const float* w1  = (const float*)d_in[12];
  const float* b1  = (const float*)d_in[13];
  const float* w2  = (const float*)d_in[14];
  const float* b2  = (const float*)d_in[15];
  float* out = (float*)d_out;

  char* base = (char*)d_ws;
  size_t off = 0;
  auto alloc = [&](size_t bytes) -> void* {
    void* p = (void*)(base + off);
    off = (off + bytes + 255) & ~(size_t)255;
    return p;
  };
  const size_t QKV_E = (size_t)B_ * H_ * N_ * HD_;      // 3,145,728
  const size_t ATT_E = (size_t)B_ * H_ * N_ * N_;       // 33,554,432
  const size_t XND_E = (size_t)B_ * N_ * D_;            // 3,145,728

  u16*    q_bf   = (u16*)alloc(QKV_E * 2);
  u16*    k_bf   = (u16*)alloc(QKV_E * 2);
  u16*    vT_bf  = (u16*)alloc(QKV_E * 2);
  u16*    attn   = (u16*)alloc(ATT_E * 2);
  float*  r1     = (float*)alloc(XND_E * 4);
  float*  x1f    = (float*)alloc(XND_E * 4);
  u16*    pw_bf  = (u16*)alloc((size_t)D_ * D_ * 2);
  u16*    w1_bf  = (u16*)alloc((size_t)HID_ * D_ * 2);
  u16*    w2_bf  = (u16*)alloc((size_t)D_ * HID_ * 2);
  float*  colsum = (float*)alloc((size_t)B_ * H_ * HD_ * 4);
  double* stats  = (double*)alloc(64 * 8);
  float*  coef   = (float*)alloc(64 * 4);
  // dead-buffer aliases
  u16* ctx = q_bf;            // context, written after q last read
  u16* x1b = k_bf;            // bf16 x1, written after k last read
  u16* hb  = (u16*)attn;      // FF hidden, written after attn last read
  float* r2 = r1;             // second residual reuses r1

  k_zero_stats<<<1, 64, 0, stream>>>(stats);
  k_cvt<<<(D_ * D_) / 256, 256, 0, stream>>>(pw, pw_bf, D_ * D_);
  k_cvt<<<(HID_ * D_) / 256, 256, 0, stream>>>(w1, w1_bf, HID_ * D_);
  k_cvt<<<(D_ * HID_) / 256, 256, 0, stream>>>(w2, w2_bf, D_ * HID_);

  k_qkv_conv<<<B_ * N_, 256, 0, stream>>>(x, qw, kw, vw, q_bf, k_bf, vT_bf);
  k_colsum<<<(B_ * H_ * HD_) / 8, 256, 0, stream>>>(vT_bf, colsum);
  k_qk_softmax<<<B_ * H_ * (N_ / 16), 256, 0, stream>>>(q_bf, k_bf, attn);
  k_mix_stats<<<(B_ * N_ * N_) / 256, 256, 0, stream>>>(attn, rw, rb, stats);
  k_bn_finalize<<<1, 8, 0, stream>>>(stats, bng, bnb, coef);
  k_attnv<<<B_ * H_ * (N_ / 128), 256, 0, stream>>>(attn, vT_bf, coef, colsum, ctx);

  k_proj<<<dim3((B_ * N_) / 64, D_ / 64), 256, 0, stream>>>(ctx, pw_bf, pb, x, r1, stats);
  k_ln_finalize<<<1, 4, 0, stream>>>(stats + 16, stats + 20, coef + 16);
  k_ln1_apply<<<(int)(XND_E / 256), 256, 0, stream>>>(r1, coef + 16, lng, lnb, x1f, x1b);

  k_ff1<<<dim3((B_ * N_) / 64, HID_ / 64), 256, 0, stream>>>(x1b, w1_bf, b1, hb);
  k_ff2<<<dim3((B_ * N_) / 64, D_ / 64), 256, 0, stream>>>(hb, w2_bf, b2, x1f, r2, stats);
  k_ln_finalize<<<1, 4, 0, stream>>>(stats + 24, stats + 28, coef + 24);
  k_ln2_apply<<<(int)(XND_E / 256), 256, 0, stream>>>(r2, coef + 24, lng, lnb, out);
}
